// EnhancedSpatiotemporalModel_33887291965958
// MI455X (gfx1250) — matrix operand order repaired and checked by reading, not compiled, not run on hardware
//
#include <hip/hip_runtime.h>
#include <hip/hip_bf16.h>
#include <math.h>

// ---------------------------------------------------------------------------
// Model dims
// ---------------------------------------------------------------------------
#define BB   256
#define TT   365
#define TP   368          // T padded to multiple of 16
#define SD   64
#define DD   16
#define HH   256
#define NHD  8
#define HDM  32
#define LL   4
#define FFD  1024
#define KNB  5
#define NST  10000
#define PP   30
#define CIN  384          // SD*(K+1)

typedef __attribute__((ext_vector_type(16))) _Float16 v16h;
typedef __attribute__((ext_vector_type(8)))  float    v8f;

union FragA { unsigned u[8]; v16h h; };

__device__ __forceinline__ float gelu_f(float v) {
    return 0.5f * v * (1.f + erff(v * 0.70710678118654752f));
}

// ---------------------------------------------------------------------------
// Generic WMMA GEMM:  C[M,N] = act(A[M,K] @ Bw[K,N] + bias)
// Block: 128 threads (4 waves). Block tile 128(M) x 64(N), K step 32.
// Each wave owns a 32x64 strip: 2 A-fragments x 4 B-fragments = 8 WMMAs per
// K-step. A/B staged in LDS as f16 (A m-major stride 32, B n-major stride 32)
// so WMMA fragment K-pairs are contiguous 32-bit LDS reads (merge to b128).
// Next K-step tiles are prefetched with global_prefetch.
// EPI: 0 f32 out, 1 f16 out, 2 relu->f16, 3 gelu->f32, 4 tanh*scale+shift->f32
// ---------------------------------------------------------------------------
template<int EPI, bool AHALF>
__global__ __launch_bounds__(128)
void gemm_wmma(const void* __restrict__ Av, const float* __restrict__ Bw,
               const float* __restrict__ bias, void* __restrict__ Cout,
               int M, int N, int Kd,
               const float* __restrict__ scale, const float* __restrict__ shift)
{
    __shared__ _Float16 As[128 * 32];
    __shared__ _Float16 Bs[64 * 32];

    const int t  = threadIdx.x;
    const int w  = t >> 5;          // wave id 0..3 -> 32-row M sub-strip
    const int l  = t & 31;          // lane
    const int n0 = blockIdx.x * 64;
    const int m0 = blockIdx.y * 128;

    v8f acc[2][4];
#pragma unroll
    for (int mt = 0; mt < 2; ++mt)
#pragma unroll
        for (int nt = 0; nt < 4; ++nt)
#pragma unroll
            for (int e = 0; e < 8; ++e) acc[mt][nt][e] = 0.f;

    for (int k0 = 0; k0 < Kd; k0 += 32) {
        // ---- stage A tile (128x32), convert to f16 ----
#pragma unroll
        for (int i = 0; i < 32; ++i) {
            int lin = t + 128 * i;
            int r = lin >> 5, c = lin & 31;
            float v = 0.f;
            int gm = m0 + r, gk = k0 + c;
            if (gm < M && gk < Kd) {
                if (AHALF) v = (float)((const _Float16*)Av)[(size_t)gm * Kd + gk];
                else       v = ((const float*)Av)[(size_t)gm * Kd + gk];
            }
            As[r * 32 + c] = (_Float16)v;
        }
        // ---- stage B tile (32x64) n-major ----
#pragma unroll
        for (int i = 0; i < 16; ++i) {
            int lin = t + 128 * i;
            int n = lin & 63, k = lin >> 6;
            float v = 0.f;
            int gn = n0 + n, gk = k0 + k;
            if (gn < N && gk < Kd) v = Bw[(size_t)gk * N + gn];
            Bs[n * 32 + k] = (_Float16)v;
        }
        // ---- prefetch next K-step tiles (global_prefetch_b8) ----
        if (k0 + 32 < Kd) {
            int gm = m0 + t;
            if (gm < M) {
                if (AHALF) __builtin_prefetch(&((const _Float16*)Av)[(size_t)gm * Kd + k0 + 32], 0, 3);
                else       __builtin_prefetch(&((const float*)Av)[(size_t)gm * Kd + k0 + 32], 0, 3);
            }
            if (t < 32 && (k0 + 32 + t) < Kd)
                __builtin_prefetch(&Bw[(size_t)(k0 + 32 + t) * N + n0], 0, 3);
        }
        __syncthreads();

        // ---- A fragments (two 16x32 tiles) per ISA layout ----
        FragA fa[2];
#pragma unroll
        for (int mt = 0; mt < 2; ++mt) {
            int m    = (l & 15) + w * 32 + mt * 16;
            int koff = (l >> 4) * 8;
#pragma unroll
            for (int r = 0; r < 8; ++r) {
                int kk = koff + 2 * r + ((r >= 4) ? 8 : 0);
                fa[mt].u[r] = *(const unsigned*)&As[m * 32 + kk];
            }
        }
        // ---- 4 N-tiles, 2 WMMAs each ----
#pragma unroll
        for (int nt = 0; nt < 4; ++nt) {
            FragA fb;
            int nn   = nt * 16 + (l & 15);
            int koff = (l >> 4) * 8;
#pragma unroll
            for (int r = 0; r < 8; ++r)
                fb.u[r] = *(const unsigned*)&Bs[nn * 32 + koff + 2 * r + ((r >= 4) ? 8 : 0)];
            acc[0][nt] = __builtin_amdgcn_wmma_f32_16x16x32_f16(
                false, fa[0].h, false, fb.h, (short)0, acc[0][nt], false, false);
            acc[1][nt] = __builtin_amdgcn_wmma_f32_16x16x32_f16(
                false, fa[1].h, false, fb.h, (short)0, acc[1][nt], false, false);
        }
        __syncthreads();
    }

    // ---- epilogue ----
#pragma unroll
    for (int mt = 0; mt < 2; ++mt) {
        const int mrow0 = m0 + w * 32 + mt * 16 + ((l >> 4) * 8);
#pragma unroll
        for (int nt = 0; nt < 4; ++nt) {
            int n = n0 + nt * 16 + (l & 15);
            if (n >= N) continue;
            float bv = bias ? bias[n] : 0.f;
#pragma unroll
            for (int e = 0; e < 8; ++e) {
                int m = mrow0 + e;
                if (m >= M) continue;
                float v = acc[mt][nt][e] + bv;
                size_t idx = (size_t)m * N + n;
                if (EPI == 0)      ((float*)Cout)[idx] = v;
                else if (EPI == 1) ((_Float16*)Cout)[idx] = (_Float16)v;
                else if (EPI == 2) ((_Float16*)Cout)[idx] = (_Float16)(v > 0.f ? v : 0.f);
                else if (EPI == 3) ((float*)Cout)[idx] = gelu_f(v);
                else               ((float*)Cout)[idx] = tanhf(v) * scale[0] + shift[0];
            }
        }
    }
}

// ---------------------------------------------------------------------------
// Flash attention, one block per (b, head). 128 threads = 4 independent waves,
// each wave owns distinct 16-row query tiles. K and V^T cached in LDS.
// qkv layout: (B, TP, 3H) f16, q at [0,H), k at [H,2H), v at [2H,3H).
// ---------------------------------------------------------------------------
__global__ __launch_bounds__(128)
void attn_kernel(const _Float16* __restrict__ qkv, float* __restrict__ out)
{
    __shared__ _Float16 Ks[TP * 32];       // [key][d]
    __shared__ _Float16 Vst[32 * 384];     // [d][key], zero-padded keys 368..383
    __shared__ _Float16 Pbuf[4][16 * 32];  // per-wave prob tile (16 q x 32 keys)

    const int b = blockIdx.x >> 3;
    const int h = blockIdx.x & 7;
    const int t = threadIdx.x;
    const int w = t >> 5;
    const int l = t & 31;
    const unsigned* qkvu = (const unsigned*)qkv;

    for (int lin = t; lin < TP * 32; lin += 128) {
        int tt = lin >> 5, d = lin & 31;
        Ks[lin] = qkv[((size_t)(b * TP + tt)) * 768 + 256 + h * 32 + d];
    }
    for (int lin = t; lin < 32 * 384; lin += 128) {
        int tt = lin % 384, d = lin / 384;
        _Float16 v = (_Float16)0.f;
        if (tt < TP) v = qkv[((size_t)(b * TP + tt)) * 768 + 512 + h * 32 + d];
        Vst[d * 384 + tt] = v;
    }
    __syncthreads();

    const float inv_s = 0.17677669529663687f;   // 1/sqrt(32)
    _Float16* Pw = Pbuf[w];

    for (int qt = w; qt < TP / 16; qt += 4) {
        const int t0 = qt * 16;
        // q A-fragment straight from global (already f16)
        FragA fq;
        {
            int m    = l & 15;
            int koff = (l >> 4) * 8;
            size_t base = ((size_t)(b * TP + t0 + m)) * 768 + (size_t)h * 32;
#pragma unroll
            for (int r = 0; r < 8; ++r) {
                int d = koff + 2 * r + ((r >= 4) ? 8 : 0);
                fq.u[r] = qkvu[(base + d) >> 1];
            }
        }

        v8f acc0, acc1;
        float mrun[8], lrun[8];
#pragma unroll
        for (int e = 0; e < 8; ++e) { acc0[e] = 0.f; acc1[e] = 0.f; mrun[e] = -1e30f; lrun[e] = 0.f; }

        for (int kt = 0; kt < TP / 16; kt += 2) {
            v8f s0, s1;
#pragma unroll
            for (int e = 0; e < 8; ++e) { s0[e] = 0.f; s1[e] = 0.f; }
            {   // scores tile kt : S = q @ K^T
                FragA fk;
                int key  = kt * 16 + (l & 15);
                int koff = (l >> 4) * 8;
#pragma unroll
                for (int r = 0; r < 8; ++r)
                    fk.u[r] = *(const unsigned*)&Ks[key * 32 + koff + 2 * r + ((r >= 4) ? 8 : 0)];
                s0 = __builtin_amdgcn_wmma_f32_16x16x32_f16(
                    false, fq.h, false, fk.h, (short)0, s0, false, false);
            }
            const bool has1 = (kt + 1) < TP / 16;
            if (has1) {
                FragA fk;
                int key  = (kt + 1) * 16 + (l & 15);
                int koff = (l >> 4) * 8;
#pragma unroll
                for (int r = 0; r < 8; ++r)
                    fk.u[r] = *(const unsigned*)&Ks[key * 32 + koff + 2 * r + ((r >= 4) ? 8 : 0)];
                s1 = __builtin_amdgcn_wmma_f32_16x16x32_f16(
                    false, fq.h, false, fk.h, (short)0, s1, false, false);
            }
            const int c0 = kt * 16 + (l & 15);
            const int c1 = (kt + 1) * 16 + (l & 15);
#pragma unroll
            for (int e = 0; e < 8; ++e) {
                s0[e] = (c0 < TT) ? s0[e] * inv_s : -1e30f;
                s1[e] = (has1 && c1 < TT) ? s1[e] * inv_s : -1e30f;
            }
            // online softmax per q row (row-wise reduce across the 16 n-lanes)
#pragma unroll
            for (int e = 0; e < 8; ++e) {
                float tmax = fmaxf(s0[e], s1[e]);
#pragma unroll
                for (int msk = 1; msk < 16; msk <<= 1)
                    tmax = fmaxf(tmax, __shfl_xor(tmax, msk, 32));
                float newm = fmaxf(mrun[e], tmax);
                float corr = expf(mrun[e] - newm);
                float p0 = expf(s0[e] - newm);
                float p1 = expf(s1[e] - newm);
                float rs = p0 + p1;
#pragma unroll
                for (int msk = 1; msk < 16; msk <<= 1)
                    rs += __shfl_xor(rs, msk, 32);
                lrun[e] = lrun[e] * corr + rs;
                mrun[e] = newm;
                acc0[e] *= corr; acc1[e] *= corr;
                int row = e + ((l >> 4) * 8);
                Pw[row * 32 + (l & 15)]      = (_Float16)p0;
                Pw[row * 32 + 16 + (l & 15)] = (_Float16)p1;
            }
            asm volatile("s_wait_dscnt 0" ::: "memory");   // C-layout -> A-layout bounce
            // probabilities as A-fragment
            FragA fp;
            {
                int m    = l & 15;
                int koff = (l >> 4) * 8;
#pragma unroll
                for (int r = 0; r < 8; ++r) {
                    int d = koff + 2 * r + ((r >= 4) ? 8 : 0);
                    fp.u[r] = *(const unsigned*)&Pw[m * 32 + d];
                }
            }
            // V B-fragments (32 keys x 16 dims, two dim halves)
            FragA fv0, fv1;
            {
                int koff = (l >> 4) * 8;
                int d0 = l & 15, d1 = 16 + (l & 15);
#pragma unroll
                for (int r = 0; r < 8; ++r) {
                    int key = kt * 16 + koff + 2 * r + ((r >= 4) ? 8 : 0);
                    fv0.u[r] = *(const unsigned*)&Vst[d0 * 384 + key];
                    fv1.u[r] = *(const unsigned*)&Vst[d1 * 384 + key];
                }
            }
            acc0 = __builtin_amdgcn_wmma_f32_16x16x32_f16(
                false, fp.h, false, fv0.h, (short)0, acc0, false, false);
            acc1 = __builtin_amdgcn_wmma_f32_16x16x32_f16(
                false, fp.h, false, fv1.h, (short)0, acc1, false, false);
        }
        // normalize + store
#pragma unroll
        for (int e = 0; e < 8; ++e) {
            int row = e + ((l >> 4) * 8);
            float invl = 1.f / lrun[e];
            size_t o = ((size_t)(b * TP + t0 + row)) * 256 + h * 32;
            out[o + (l & 15)]      = acc0[e] * invl;
            out[o + 16 + (l & 15)] = acc1[e] * invl;
        }
    }
}

// ---------------------------------------------------------------------------
// Haversine + top-(K+1) nearest stations, write neighbor indices 1..K
// ---------------------------------------------------------------------------
__device__ __forceinline__ bool tk_less(float d1, int i1, float d2, int i2) {
    return (d1 < d2) || (d1 == d2 && i1 < i2);
}

__global__ __launch_bounds__(128)
void topk_kernel(const float* __restrict__ lat, const float* __restrict__ lon,
                 const float* __restrict__ sc, int* __restrict__ nbr)
{
    const float DEG = 0.017453292519943295f;
    int b = blockIdx.x, t = threadIdx.x;
    float lat_o = (lat[b] + 1.f) * 0.5f * 25.f + 24.f;
    float lon_o = (lon[b] + 1.f) * 0.5f * 58.5f - 125.f;
    float qlat = lat_o * DEG, qlon = lon_o * DEG;
    float cq = cosf(qlat);

    float bd[6]; int bi[6];
#pragma unroll
    for (int i = 0; i < 6; ++i) { bd[i] = 1e30f; bi[i] = 0x7fffffff; }
    for (int n = t; n < NST; n += 128) {
        float slat = sc[2 * n], slon = sc[2 * n + 1];
        float sa = sinf((qlat - slat) * 0.5f);
        float sb = sinf((qlon - slon) * 0.5f);
        float hav = sa * sa + cq * cosf(slat) * sb * sb;
        if (tk_less(hav, n, bd[5], bi[5])) {
            int j = 5;
            while (j > 0 && tk_less(hav, n, bd[j - 1], bi[j - 1])) {
                bd[j] = bd[j - 1]; bi[j] = bi[j - 1]; --j;
            }
            bd[j] = hav; bi[j] = n;
        }
    }
    __shared__ float sd[128 * 6];
    __shared__ int   si[128 * 6];
    for (int i = 0; i < 6; ++i) { sd[t * 6 + i] = bd[i]; si[t * 6 + i] = bi[i]; }
    __syncthreads();
    if (t == 0) {
        float fd[6]; int fi[6];
#pragma unroll
        for (int i = 0; i < 6; ++i) { fd[i] = 1e30f; fi[i] = 0x7fffffff; }
        for (int c = 0; c < 128 * 6; ++c) {
            float hv = sd[c]; int ix = si[c];
            if (tk_less(hv, ix, fd[5], fi[5])) {
                int j = 5;
                while (j > 0 && tk_less(hv, ix, fd[j - 1], fi[j - 1])) {
                    fd[j] = fd[j - 1]; fi[j] = fi[j - 1]; --j;
                }
                fd[j] = hv; fi[j] = ix;
            }
        }
        for (int i = 0; i < KNB; ++i) nbr[b * KNB + i] = fi[i + 1];
    }
}

// ---------------------------------------------------------------------------
// Elementwise / LN kernels
// ---------------------------------------------------------------------------
__global__ __launch_bounds__(256)
void gather_combined(const float* __restrict__ stat, const float* __restrict__ all_st,
                     const int* __restrict__ nbr, float* __restrict__ comb)
{
    int idx = blockIdx.x * 256 + threadIdx.x;
    if (idx >= BB * CIN) return;
    int b = idx / CIN, c = idx % CIN;
    float v;
    if (c < SD) v = stat[b * SD + c];
    else {
        int j = (c - SD) / SD, cc = (c - SD) % SD;
        v = all_st[(size_t)nbr[b * KNB + j] * SD + cc];
    }
    comb[idx] = v;
}

__global__ __launch_bounds__(256)
void dyn_expand(const float* __restrict__ dyn, float* __restrict__ A0)
{
    int idx = blockIdx.x * 256 + threadIdx.x;
    if (idx >= BB * TP * DD) return;
    int b = idx / (TP * DD);
    int r = idx % (TP * DD);
    int tt = r / DD, c = r % DD;
    A0[idx] = (tt < TT) ? dyn[((size_t)(b * TT + tt)) * DD + c] : 0.f;
}

__device__ __forceinline__ float block_ln(float v, float* rs, float* rq,
                                          int t, float g, float be)
{
    rs[t] = v; rq[t] = v * v;
    __syncthreads();
    for (int s = 128; s > 0; s >>= 1) {
        if (t < s) { rs[t] += rs[t + s]; rq[t] += rq[t + s]; }
        __syncthreads();
    }
    float mu = rs[0] * (1.f / 256.f);
    float var = rq[0] * (1.f / 256.f) - mu * mu;
    __syncthreads();
    return g * (v - mu) * rsqrtf(var + 1e-5f) + be;
}

__global__ __launch_bounds__(256)
void ln_gelu_kernel(const float* __restrict__ in, const float* __restrict__ g,
                    const float* __restrict__ be, float* __restrict__ out)
{
    __shared__ float rs[256], rq[256];
    int row = blockIdx.x, t = threadIdx.x;
    float v = in[(size_t)row * 256 + t];
    out[(size_t)row * 256 + t] = gelu_f(block_ln(v, rs, rq, t, g[t], be[t]));
}

__global__ __launch_bounds__(256)
void xinit_kernel(const float* __restrict__ pre, const float* __restrict__ g,
                  const float* __restrict__ be, const float* __restrict__ sfeat,
                  float* __restrict__ x)
{
    __shared__ float rs[256], rq[256];
    int row = blockIdx.x, t = threadIdx.x;
    float v = pre[(size_t)row * 256 + t];
    float nv = gelu_f(block_ln(v, rs, rq, t, g[t], be[t]));
    x[(size_t)row * 256 + t] = nv + sfeat[(row / TP) * 256 + t];
}

__global__ __launch_bounds__(256)
void resln_kernel(float* __restrict__ x, const float* __restrict__ y,
                  const float* __restrict__ g, const float* __restrict__ be)
{
    __shared__ float rs[256], rq[256];
    int row = blockIdx.x, t = threadIdx.x;
    float v = x[(size_t)row * 256 + t] + y[(size_t)row * 256 + t];
    x[(size_t)row * 256 + t] = block_ln(v, rs, rq, t, g[t], be[t]);
}

__global__ __launch_bounds__(256)
void pool_kernel(const float* __restrict__ x, const float* __restrict__ sfeat,
                 float* __restrict__ z)
{
    int b = blockIdx.x, c = threadIdx.x;
    float s = 0.f;
    for (int tt = 0; tt < TT; ++tt) s += x[((size_t)(b * TP + tt)) * 256 + c];
    z[b * 256 + c] = s * (1.f / (float)TT) + sfeat[b * 256 + c];
}

// ---------------------------------------------------------------------------
// Host launcher
// ---------------------------------------------------------------------------
extern "C" void kernel_launch(void* const* d_in, const int* in_sizes, int n_in,
                              void* d_out, int out_size, void* d_ws, size_t ws_size,
                              hipStream_t stream)
{
    (void)in_sizes; (void)n_in; (void)out_size; (void)ws_size;

    const float* stat   = (const float*)d_in[0];
    const float* dynam  = (const float*)d_in[1];
    const float* lat    = (const float*)d_in[2];
    const float* lon    = (const float*)d_in[3];
    const float* all_st = (const float*)d_in[4];
    const float* sc     = (const float*)d_in[5];
    const float* W_st   = (const float*)d_in[6];
    const float* b_st   = (const float*)d_in[7];
    const float* g_st   = (const float*)d_in[8];
    const float* be_st  = (const float*)d_in[9];
    const float* W_dy   = (const float*)d_in[10];
    const float* b_dy   = (const float*)d_in[11];
    const float* g_dy   = (const float*)d_in[12];
    const float* be_dy  = (const float*)d_in[13];
    const float* Wqkv   = (const float*)d_in[14];
    const float* bqkv   = (const float*)d_in[15];
    const float* Wo     = (const float*)d_in[16];
    const float* bo     = (const float*)d_in[17];
    const float* g_ln1  = (const float*)d_in[18];
    const float* be_ln1 = (const float*)d_in[19];
    const float* Wf1    = (const float*)d_in[20];
    const float* bf1    = (const float*)d_in[21];
    const float* Wf2    = (const float*)d_in[22];
    const float* bf2    = (const float*)d_in[23];
    const float* g_ln2  = (const float*)d_in[24];
    const float* be_ln2 = (const float*)d_in[25];
    const float* W_o1   = (const float*)d_in[26];
    const float* b_o1   = (const float*)d_in[27];
    const float* W_o2   = (const float*)d_in[28];
    const float* b_o2   = (const float*)d_in[29];
    const float* scale  = (const float*)d_in[30];
    const float* shift  = (const float*)d_in[31];
    float* outp = (float*)d_out;

    // workspace carve-up (256B aligned)
    char* wsb = (char*)d_ws;
    size_t off = 0;
    auto carve = [&](size_t bytes) -> void* {
        void* p = (void*)(wsb + off);
        off = (off + bytes + 255) & ~(size_t)255;
        return p;
    };
    const size_t MR = (size_t)BB * TP;              // 94208 rows
    int*       nbr   = (int*)      carve((size_t)BB * KNB * 4);
    float*     comb  = (float*)    carve((size_t)BB * CIN * 4);
    float*     tmp1  = (float*)    carve((size_t)BB * HH * 4);    // also z
    float*     sfeat = (float*)    carve((size_t)BB * HH * 4);
    float*     ubuf  = (float*)    carve((size_t)BB * 128 * 4);
    float*     A0    = (float*)    carve(MR * DD * 4);
    float*     x     = (float*)    carve(MR * HH * 4);
    float*     y     = (float*)    carve(MR * HH * 4);
    float*     attn  = (float*)    carve(MR * HH * 4);
    _Float16*  big   = (_Float16*) carve(MR * 1024 * 2);          // qkv(768) / h1(1024)

    auto grid2 = [](int M, int N) { return dim3((unsigned)((N + 63) / 64), (unsigned)((M + 127) / 128), 1); };

    // 1) neighbors + static features
    topk_kernel<<<BB, 128, 0, stream>>>(lat, lon, sc, nbr);
    gather_combined<<<(BB * CIN + 255) / 256, 256, 0, stream>>>(stat, all_st, nbr, comb);
    gemm_wmma<0, false><<<grid2(BB, HH), 128, 0, stream>>>(comb, W_st, b_st, tmp1, BB, HH, CIN, nullptr, nullptr);
    ln_gelu_kernel<<<BB, 256, 0, stream>>>(tmp1, g_st, be_st, sfeat);

    // 2) dynamic embedding -> x
    dyn_expand<<<(int)((MR * DD + 255) / 256), 256, 0, stream>>>(dynam, A0);
    gemm_wmma<0, false><<<grid2((int)MR, HH), 128, 0, stream>>>(A0, W_dy, b_dy, y, (int)MR, HH, DD, nullptr, nullptr);
    xinit_kernel<<<(int)MR, 256, 0, stream>>>(y, g_dy, be_dy, sfeat, x);

    // 3) transformer layers
    for (int L = 0; L < LL; ++L) {
        const float* Wqkv_l = Wqkv + (size_t)L * HH * 3 * HH;
        const float* bqkv_l = bqkv + (size_t)L * 3 * HH;
        const float* Wo_l   = Wo   + (size_t)L * HH * HH;
        const float* bo_l   = bo   + (size_t)L * HH;
        const float* Wf1_l  = Wf1  + (size_t)L * HH * FFD;
        const float* bf1_l  = bf1  + (size_t)L * FFD;
        const float* Wf2_l  = Wf2  + (size_t)L * FFD * HH;
        const float* bf2_l  = bf2  + (size_t)L * HH;

        gemm_wmma<1, false><<<grid2((int)MR, 3 * HH), 128, 0, stream>>>(x, Wqkv_l, bqkv_l, big, (int)MR, 3 * HH, HH, nullptr, nullptr);
        attn_kernel<<<BB * NHD, 128, 0, stream>>>(big, attn);
        gemm_wmma<0, false><<<grid2((int)MR, HH), 128, 0, stream>>>(attn, Wo_l, bo_l, y, (int)MR, HH, HH, nullptr, nullptr);
        resln_kernel<<<(int)MR, 256, 0, stream>>>(x, y, g_ln1 + (size_t)L * HH, be_ln1 + (size_t)L * HH);
        gemm_wmma<2, false><<<grid2((int)MR, FFD), 128, 0, stream>>>(x, Wf1_l, bf1_l, big, (int)MR, FFD, HH, nullptr, nullptr);
        gemm_wmma<0, true><<<grid2((int)MR, HH), 128, 0, stream>>>(big, Wf2_l, bf2_l, y, (int)MR, HH, FFD, nullptr, nullptr);
        resln_kernel<<<(int)MR, 256, 0, stream>>>(x, y, g_ln2 + (size_t)L * HH, be_ln2 + (size_t)L * HH);
    }

    // 4) head
    pool_kernel<<<BB, 256, 0, stream>>>(x, sfeat, tmp1);
    gemm_wmma<3, false><<<grid2(BB, 128), 128, 0, stream>>>(tmp1, W_o1, b_o1, ubuf, BB, 128, HH, nullptr, nullptr);
    gemm_wmma<4, false><<<grid2(BB, PP), 128, 0, stream>>>(ubuf, W_o2, b_o2, outp, BB, PP, 128, scale, shift);
}
